// QViTAttention_27273042330106
// MI455X (gfx1250) — matrix operand order repaired and checked by reading, not compiled, not run on hardware
//
#include <hip/hip_runtime.h>

typedef __attribute__((ext_vector_type(16))) _Float16 v16h;
typedef __attribute__((ext_vector_type(8)))  float    v8f;

// ---------------------------------------------------------------- constants
#define BB   32
#define NN   577
#define NP   608          // N padded to multiple of 32
#define CC   1024
#define HH   16
#define DD   64
#define MTOT (BB*NN)      // 18464
#define QMAXF 255.0f

// ---------------------------------------------------------------- helpers
__device__ inline unsigned fmap(float f) {
  unsigned u = __float_as_uint(f);
  return u ^ ((u >> 31) ? 0xFFFFFFFFu : 0x80000000u);
}
__device__ inline float funmap(unsigned u) {
  unsigned b = (u & 0x80000000u) ? (u ^ 0x80000000u) : ~u;
  return __uint_as_float(b);
}
__device__ inline float fq_apply(float v, float s, float zp) {
  float q = fminf(fmaxf(rintf(v / s) + zp, 0.0f), QMAXF);
  return (q - zp) * s;
}

__device__ inline v8f wmma_f16(v16h a, v16h b, v8f c) {
  return __builtin_amdgcn_wmma_f32_16x16x32_f16(
      false, a, false, b, (short)0, c, false, false);
}

// Async 16-byte Global->LDS copy (CDNA5 GLOBAL_LOAD_ASYNC_TO_LDS_B128, GV mode).
// LDS dest byte address = low 32 bits of the generic pointer (flat aperture rule).
__device__ inline void async_ld_b128(void* lds, const void* g) {
  unsigned ldsoff = (unsigned)(size_t)lds;
  asm volatile("global_load_async_to_lds_b128 %0, %1, off"
               :: "v"(ldsoff), "v"(g)
               : "memory");
}
__device__ inline void wait_async0() {
  asm volatile("s_wait_asynccnt 0x0" ::: "memory");
}

// A fragment: 16 rows x 32 K, element layout per CDNA5 ISA 7.12.2
__device__ inline v16h frag_a(const _Float16* base, int stride, int lane) {
  int grp = (lane >> 4) & 1;
  int r   = lane & 15;
  const _Float16* rp = base + r * stride;
  v16h a;
#pragma unroll
  for (int j = 0; j < 4; ++j) {
    int k0 = grp * 8 + 2 * j;
    a[2 * j]     = rp[k0];
    a[2 * j + 1] = rp[k0 + 1];
    int k1 = 16 + grp * 8 + 2 * j;
    a[8 + 2 * j]     = rp[k1];
    a[8 + 2 * j + 1] = rp[k1 + 1];
  }
  return a;
}

// B fragment: 32 K x 16 N; element (k,n) fetched at base + k*sk + n*sn
__device__ inline v16h frag_b(const _Float16* base, int sk, int sn, int lane) {
  int grp = (lane >> 4) & 1;
  int n   = lane & 15;
  const _Float16* p = base + n * sn + grp * 8 * sk;
  v16h b;
#pragma unroll
  for (int i = 0; i < 16; ++i) b[i] = p[(i + (i >= 8 ? 8 : 0)) * sk];
  return b;
}

// ---------------------------------------------------------------- init
__global__ void __launch_bounds__(256) k_zero_u32(unsigned* p, size_t n) {
  for (size_t i = (size_t)blockIdx.x * blockDim.x + threadIdx.x; i < n;
       i += (size_t)gridDim.x * blockDim.x) p[i] = 0u;
}
__global__ void k_init_mm(unsigned* mm) {
  int t = threadIdx.x;
  if (t < 4) { mm[2 * t] = 0xFFFFFFFFu; mm[2 * t + 1] = 0u; }
}

// ------------------------------------------------- per-channel x quant params
__global__ void __launch_bounds__(256) k_xchan(const float* __restrict__ x,
                                               float* __restrict__ sx,
                                               float* __restrict__ zpx) {
  int c = blockIdx.x, tid = threadIdx.x;
  float lmn = 3.0e38f, lmx = -3.0e38f;
  for (int m = tid; m < MTOT; m += 256) {
    float v = x[(size_t)m * CC + c];
    lmn = fminf(lmn, v); lmx = fmaxf(lmx, v);
  }
  __shared__ float smn[256], smx[256];
  smn[tid] = lmn; smx[tid] = lmx;
  __syncthreads();
  for (int s = 128; s > 0; s >>= 1) {
    if (tid < s) {
      smn[tid] = fminf(smn[tid], smn[tid + s]);
      smx[tid] = fmaxf(smx[tid], smx[tid + s]);
    }
    __syncthreads();
  }
  if (tid == 0) {
    float mn = fminf(smn[0], 0.0f), mx = fmaxf(smx[0], 0.0f);
    float s = fmaxf((mx - mn) * (1.0f / QMAXF), 1e-8f);
    sx[c] = s; zpx[c] = rintf(-mn / s);
  }
}

__global__ void __launch_bounds__(256) k_quant_x(const float* __restrict__ x,
                                                 const float* __restrict__ sx,
                                                 const float* __restrict__ zpx,
                                                 _Float16* __restrict__ xh) {
  const size_t n = (size_t)MTOT * CC;
  for (size_t i = (size_t)blockIdx.x * blockDim.x + threadIdx.x; i < n;
       i += (size_t)gridDim.x * blockDim.x) {
    int c = (int)(i & (CC - 1));
    xh[i] = (_Float16)fq_apply(x[i], sx[c], zpx[c]);
  }
}

// ---------------------------------------------- per-row weight quant -> f16
__global__ void __launch_bounds__(256) k_quant_w(const float* __restrict__ w,
                                                 _Float16* __restrict__ wh) {
  int r = blockIdx.x, tid = threadIdx.x;
  const float* row = w + (size_t)r * CC;
  float lmn = 3.0e38f, lmx = -3.0e38f;
  for (int c = tid; c < CC; c += 256) {
    float v = row[c];
    lmn = fminf(lmn, v); lmx = fmaxf(lmx, v);
  }
  __shared__ float smn[256], smx[256];
  __shared__ float sS, sZ;
  smn[tid] = lmn; smx[tid] = lmx;
  __syncthreads();
  for (int s = 128; s > 0; s >>= 1) {
    if (tid < s) {
      smn[tid] = fminf(smn[tid], smn[tid + s]);
      smx[tid] = fmaxf(smx[tid], smx[tid + s]);
    }
    __syncthreads();
  }
  if (tid == 0) {
    float mn = fminf(smn[0], 0.0f), mx = fmaxf(smx[0], 0.0f);
    float s = fmaxf((mx - mn) * (1.0f / QMAXF), 1e-8f);
    sS = s; sZ = rintf(-mn / s);
  }
  __syncthreads();
  for (int c = tid; c < CC; c += 256)
    wh[(size_t)r * CC + c] = (_Float16)fq_apply(row[c], sS, sZ);
}

// ---------------------------------------------------------------- WMMA GEMM
// out = A[M x 1024] * W[Nout x 1024]^T + bias
// MODE 0: write f32 row-major [M x Nout]
// MODE 1: qkv epilogue -> f16 padded [3][B][H][NP][DD]
// Tiles staged via async Global->LDS with double buffering.
__device__ inline void gemm_issue_tile(const _Float16* __restrict__ A,
                                       const _Float16* __restrict__ W,
                                       int m0, int n0, int kt, int M,
                                       _Float16 (*As)[40], _Float16 (*Ws)[40],
                                       int tid) {
  // A tile: 128 rows x 32 halves = 512 x 16B chunks, 2 per thread
#pragma unroll
  for (int q = 0; q < 2; ++q) {
    int c = tid + q * 256;
    int r = c >> 2, cp = c & 3;      // 4 chunks (8 halves each) per row
    int gm = m0 + r;
    if (gm < M)
      async_ld_b128(&As[r][cp * 8], A + (size_t)gm * CC + kt + cp * 8);
  }
  // W tile: 64 rows x 32 halves = 256 chunks, 1 per thread
  {
    int r = tid >> 2, cp = tid & 3;
    async_ld_b128(&Ws[r][cp * 8], W + (size_t)(n0 + r) * CC + kt + cp * 8);
  }
}

template <int MODE>
__global__ void __launch_bounds__(256) k_gemm(const _Float16* __restrict__ A,
                                              const _Float16* __restrict__ W,
                                              const float* __restrict__ bias,
                                              float* __restrict__ outF,
                                              _Float16* __restrict__ outH,
                                              int M, int Nout) {
  __shared__ _Float16 As[2][128][40];
  __shared__ _Float16 Ws[2][64][40];
  int tid = threadIdx.x, lane = tid & 31, wid = tid >> 5;
  int wm = wid & 3, wn = wid >> 2;
  int m0 = blockIdx.x * 128, n0 = blockIdx.y * 64;

  v8f acc[2][2] = {};

  gemm_issue_tile(A, W, m0, n0, 0, M, As[0], Ws[0], tid);

  for (int kt = 0; kt < CC / 32; ++kt) {
    wait_async0();          // tile kt landed (per-wave), then block-wide sync
    __syncthreads();
    if (kt + 1 < CC / 32)   // stream tile kt+1 while computing on tile kt
      gemm_issue_tile(A, W, m0, n0, (kt + 1) * 32, M,
                      As[(kt + 1) & 1], Ws[(kt + 1) & 1], tid);

    int cb = kt & 1;
    v16h a0 = frag_a(&As[cb][wm * 32][0], 40, lane);
    v16h a1 = frag_a(&As[cb][wm * 32 + 16][0], 40, lane);
    v16h b0 = frag_b(&Ws[cb][wn * 32][0], 1, 40, lane);
    v16h b1 = frag_b(&Ws[cb][wn * 32 + 16][0], 1, 40, lane);
    acc[0][0] = wmma_f16(a0, b0, acc[0][0]);
    acc[0][1] = wmma_f16(a0, b1, acc[0][1]);
    acc[1][0] = wmma_f16(a1, b0, acc[1][0]);
    acc[1][1] = wmma_f16(a1, b1, acc[1][1]);
    __syncthreads();        // all waves done reading buffer cb
  }

  int grp = lane >> 4, nn = lane & 15;
#pragma unroll
  for (int sm = 0; sm < 2; ++sm)
#pragma unroll
    for (int sn = 0; sn < 2; ++sn)
#pragma unroll
      for (int r = 0; r < 8; ++r) {
        int row = wm * 32 + sm * 16 + grp * 8 + r;
        int col = wn * 32 + sn * 16 + nn;
        int gm = m0 + row, gj = n0 + col;
        if (gm < M) {
          float v = acc[sm][sn][r] + bias[gj];
          if (MODE == 0) {
            outF[(size_t)gm * Nout + gj] = v;
          } else {
            int t = gj >> 10, rem = gj & 1023;
            int h = rem >> 6, d = rem & 63;
            int b = gm / NN, n = gm - b * NN;
            size_t off = ((((size_t)t * BB + b) * HH + h) * NP + n) * DD + d;
            outH[off] = (_Float16)v;
          }
        }
      }
}

// ------------------------------------------------- per-tensor minmax (q,k,v)
__global__ void __launch_bounds__(256) k_minmax_qkv(const _Float16* __restrict__ p,
                                                    unsigned* __restrict__ mm) {
  const size_t per = (size_t)BB * HH * NP * DD;
  int t = blockIdx.y;
  const _Float16* q = p + (size_t)t * per;
  float lmn = 3.0e38f, lmx = -3.0e38f;
  for (size_t i = (size_t)blockIdx.x * blockDim.x + threadIdx.x; i < per;
       i += (size_t)gridDim.x * blockDim.x) {
    float v = (float)q[i];
    lmn = fminf(lmn, v); lmx = fmaxf(lmx, v);
  }
#pragma unroll
  for (int off = 16; off > 0; off >>= 1) {
    lmn = fminf(lmn, __shfl_xor(lmn, off, 32));
    lmx = fmaxf(lmx, __shfl_xor(lmx, off, 32));
  }
  __shared__ float smn[8], smx[8];
  int wid = threadIdx.x >> 5, lane = threadIdx.x & 31;
  if (lane == 0) { smn[wid] = lmn; smx[wid] = lmx; }
  __syncthreads();
  if (threadIdx.x == 0) {
    float a = smn[0], b = smx[0];
    for (int i = 1; i < 8; ++i) { a = fminf(a, smn[i]); b = fmaxf(b, smx[i]); }
    atomicMin(&mm[2 * t], fmap(a));
    atomicMax(&mm[2 * t + 1], fmap(b));
  }
}

__global__ void __launch_bounds__(256) k_minmax_f32(const float* __restrict__ p,
                                                    size_t n,
                                                    unsigned* __restrict__ mm) {
  float lmn = 3.0e38f, lmx = -3.0e38f;
  for (size_t i = (size_t)blockIdx.x * blockDim.x + threadIdx.x; i < n;
       i += (size_t)gridDim.x * blockDim.x) {
    float v = p[i];
    lmn = fminf(lmn, v); lmx = fmaxf(lmx, v);
  }
#pragma unroll
  for (int off = 16; off > 0; off >>= 1) {
    lmn = fminf(lmn, __shfl_xor(lmn, off, 32));
    lmx = fmaxf(lmx, __shfl_xor(lmx, off, 32));
  }
  __shared__ float smn[8], smx[8];
  int wid = threadIdx.x >> 5, lane = threadIdx.x & 31;
  if (lane == 0) { smn[wid] = lmn; smx[wid] = lmx; }
  __syncthreads();
  if (threadIdx.x == 0) {
    float a = smn[0], b = smx[0];
    for (int i = 1; i < 8; ++i) { a = fminf(a, smn[i]); b = fmaxf(b, smx[i]); }
    atomicMin(&mm[0], fmap(a));
    atomicMax(&mm[1], fmap(b));
  }
}

__global__ void k_tensor_params(const unsigned* __restrict__ mm,
                                float* __restrict__ tp) {
  int s = threadIdx.x;
  if (s < 4) {
    float mn = fminf(funmap(mm[2 * s]), 0.0f);
    float mx = fmaxf(funmap(mm[2 * s + 1]), 0.0f);
    float sc = fmaxf((mx - mn) * (1.0f / QMAXF), 1e-8f);
    tp[2 * s] = sc;
    tp[2 * s + 1] = rintf(-mn / sc);
  }
}

__global__ void __launch_bounds__(256) k_quant_qkv(_Float16* __restrict__ qkvh,
                                                   const float* __restrict__ tp) {
  const size_t per = (size_t)BB * HH * NP * DD;
  const size_t n = 3 * per;
  for (size_t i = (size_t)blockIdx.x * blockDim.x + threadIdx.x; i < n;
       i += (size_t)gridDim.x * blockDim.x) {
    int t = (int)(i / per);
    qkvh[i] = (_Float16)fq_apply((float)qkvh[i], tp[2 * t], tp[2 * t + 1]);
  }
}

__global__ void __launch_bounds__(256) k_quant_o(const float* __restrict__ ao,
                                                 const float* __restrict__ tp,
                                                 _Float16* __restrict__ oh) {
  const size_t n = (size_t)MTOT * CC;
  float s = tp[6], zp = tp[7];
  for (size_t i = (size_t)blockIdx.x * blockDim.x + threadIdx.x; i < n;
       i += (size_t)gridDim.x * blockDim.x)
    oh[i] = (_Float16)fq_apply(ao[i], s, zp);
}

// ---------------------------------------------------------------- attention
// grid: (19 row-tiles of 32, 512 (b,h)); block: 64 threads = 2 waves,
// each wave owns 16 query rows.  Q/K/V tiles staged with async-to-LDS.
__device__ inline void attn_issue_32x64(void* ldsBase, int ldsStride,
                                        const _Float16* gsrc, int row0, int tid) {
  _Float16 (*L)[72] = (_Float16 (*)[72])ldsBase;
  (void)ldsStride;
#pragma unroll
  for (int q = 0; q < 4; ++q) {         // 32 rows x 8 chunks = 256, 64 threads
    int c = tid + q * 64;
    int r = c >> 3, cp = c & 7;
    async_ld_b128(&L[r][cp * 8], gsrc + (size_t)(row0 + r) * DD + cp * 8);
  }
}

__global__ void __launch_bounds__(64) k_attn(const _Float16* __restrict__ qkvh,
                                             float* __restrict__ ao) {
  __shared__ _Float16 Sls[32][640];
  __shared__ _Float16 Qls[32][72];
  __shared__ _Float16 KVls[32][72];

  int bh = blockIdx.y;
  int b = bh >> 4, h = bh & 15;
  int n0 = blockIdx.x * 32;
  int tid = threadIdx.x, lane = tid & 31, wv = tid >> 5;
  int grp = lane >> 4, nn = lane & 15;

  const size_t headSz = (size_t)NP * DD;
  const _Float16* qh = qkvh + (((size_t)0 * BB + b) * HH + h) * headSz;
  const _Float16* kh = qkvh + (((size_t)1 * BB + b) * HH + h) * headSz;
  const _Float16* vh = qkvh + (((size_t)2 * BB + b) * HH + h) * headSz;

  // async-load Q strip (rows n0..n0+31; padded rows are zero in qkvh)
  attn_issue_32x64(Qls, 72, qh, n0, tid);

  // ---- phase 1: S = (Q K^T) * 0.125, stored f16 in Sls
  for (int ct = 0; ct < NP / 32; ++ct) {
    int c0 = ct * 32;
    __syncthreads();                    // prior consumers of KVls done
    attn_issue_32x64(KVls, 72, kh, c0, tid);
    wait_async0();                      // own async ops (incl. Q on ct==0)
    __syncthreads();
    v8f s0 = {}, s1 = {};
#pragma unroll
    for (int kk = 0; kk < DD; kk += 32) {
      v16h af = frag_a(&Qls[wv * 16][kk], 72, lane);
      v16h b0 = frag_b(&KVls[0][kk], 1, 72, lane);
      v16h b1 = frag_b(&KVls[16][kk], 1, 72, lane);
      s0 = wmma_f16(af, b0, s0);
      s1 = wmma_f16(af, b1, s1);
    }
#pragma unroll
    for (int r = 0; r < 8; ++r) {
      int row = wv * 16 + grp * 8 + r;
      Sls[row][c0 + nn]      = (_Float16)(s0[r] * 0.125f);
      Sls[row][c0 + 16 + nn] = (_Float16)(s1[r] * 0.125f);
    }
  }

  // ---- softmax (wave-local: each wave owns its 16 rows of Sls)
  for (int r = 0; r < 16; ++r) {
    int row = wv * 16 + r;
    float mx = -3.0e38f;
    for (int c = lane; c < NP; c += 32) {
      float s = (c < NN) ? (float)Sls[row][c] : -3.0e38f;
      mx = fmaxf(mx, s);
    }
#pragma unroll
    for (int off = 16; off > 0; off >>= 1) mx = fmaxf(mx, __shfl_xor(mx, off, 32));
    float sum = 0.0f;
    for (int c = lane; c < NP; c += 32) {
      float e = (c < NN) ? __expf((float)Sls[row][c] - mx) : 0.0f;
      sum += e;
      Sls[row][c] = (_Float16)e;
    }
#pragma unroll
    for (int off = 16; off > 0; off >>= 1) sum += __shfl_xor(sum, off, 32);
    float inv = 1.0f / sum;
    for (int c = lane; c < NP; c += 32)
      Sls[row][c] = (_Float16)((float)Sls[row][c] * inv);
  }

  // ---- phase 2: O = P V
  v8f o0 = {}, o1 = {}, o2 = {}, o3 = {};
  for (int kt = 0; kt < NP / 32; ++kt) {
    int k0 = kt * 32;
    __syncthreads();
    attn_issue_32x64(KVls, 72, vh, k0, tid);
    wait_async0();
    __syncthreads();
    v16h af = frag_a(&Sls[wv * 16][k0], 640, lane);
    v16h b0 = frag_b(&KVls[0][0],  72, 1, lane);
    v16h b1 = frag_b(&KVls[0][16], 72, 1, lane);
    v16h b2 = frag_b(&KVls[0][32], 72, 1, lane);
    v16h b3 = frag_b(&KVls[0][48], 72, 1, lane);
    o0 = wmma_f16(af, b0, o0);
    o1 = wmma_f16(af, b1, o1);
    o2 = wmma_f16(af, b2, o2);
    o3 = wmma_f16(af, b3, o3);
  }
#pragma unroll
  for (int r = 0; r < 8; ++r) {
    int n = n0 + wv * 16 + grp * 8 + r;
    if (n < NN) {
      size_t base = ((size_t)b * NN + n) * CC + h * DD;
      ao[base + nn]      = o0[r];
      ao[base + 16 + nn] = o1[r];
      ao[base + 32 + nn] = o2[r];
      ao[base + 48 + nn] = o3[r];
    }
  }
}

// ---------------------------------------------------------------- launch
extern "C" void kernel_launch(void* const* d_in, const int* in_sizes, int n_in,
                              void* d_out, int out_size, void* d_ws, size_t ws_size,
                              hipStream_t stream) {
  (void)in_sizes; (void)n_in; (void)out_size; (void)ws_size;
  const float* x      = (const float*)d_in[0];
  const float* qkv_w  = (const float*)d_in[1];
  const float* qkv_b  = (const float*)d_in[2];
  const float* proj_w = (const float*)d_in[3];
  const float* proj_b = (const float*)d_in[4];
  float* out = (float*)d_out;

  char* ws = (char*)d_ws;
  size_t off = 0;
  auto alloc = [&](size_t bytes) {
    void* p = ws + off;
    off += (bytes + 255) & ~(size_t)255;
    return p;
  };
  _Float16* xh   = (_Float16*)alloc((size_t)MTOT * CC * 2);
  _Float16* wqh  = (_Float16*)alloc((size_t)3 * CC * CC * 2);
  _Float16* pwh  = (_Float16*)alloc((size_t)CC * CC * 2);
  _Float16* qkvh = (_Float16*)alloc((size_t)3 * BB * HH * NP * DD * 2);
  float*    ao   = (float*)   alloc((size_t)MTOT * CC * 4);
  _Float16* oh   = (_Float16*)alloc((size_t)MTOT * CC * 2);
  float*    sx   = (float*)   alloc(CC * 4);
  float*    zpx  = (float*)   alloc(CC * 4);
  unsigned* mm   = (unsigned*)alloc(8 * 4);
  float*    tp   = (float*)   alloc(8 * 4);

  const size_t qkvhU32 = (size_t)3 * BB * HH * NP * DD / 2;

  // init
  k_init_mm<<<1, 8, 0, stream>>>(mm);
  k_zero_u32<<<4096, 256, 0, stream>>>((unsigned*)qkvh, qkvhU32);

  // activation / weight fake-quant
  k_xchan<<<CC, 256, 0, stream>>>(x, sx, zpx);
  k_quant_x<<<4096, 256, 0, stream>>>(x, sx, zpx, xh);
  k_quant_w<<<3 * CC, 256, 0, stream>>>(qkv_w, wqh);
  k_quant_w<<<CC, 256, 0, stream>>>(proj_w, pwh);

  // qkv GEMM -> padded f16 q/k/v
  {
    dim3 g((MTOT + 127) / 128, (3 * CC) / 64);
    k_gemm<1><<<g, 256, 0, stream>>>(xh, wqh, qkv_b, nullptr, qkvh, MTOT, 3 * CC);
  }

  // per-tensor quant of q,k,v
  k_minmax_qkv<<<dim3(2048, 3), 256, 0, stream>>>(qkvh, mm);
  k_tensor_params<<<1, 4, 0, stream>>>(mm, tp);
  k_quant_qkv<<<4096, 256, 0, stream>>>(qkvh, tp);

  // attention
  k_attn<<<dim3(NP / 32, BB * HH), 64, 0, stream>>>(qkvh, ao);

  // per-tensor quant of attention output
  k_minmax_f32<<<4096, 256, 0, stream>>>(ao, (size_t)MTOT * CC, mm + 6);
  k_tensor_params<<<1, 4, 0, stream>>>(mm, tp);
  k_quant_o<<<4096, 256, 0, stream>>>(ao, tp, oh);

  // proj GEMM -> d_out (f32)
  {
    dim3 g((MTOT + 127) / 128, CC / 64);
    k_gemm<0><<<g, 256, 0, stream>>>(oh, pwh, proj_b, out, nullptr, MTOT, CC);
  }
}
